// Model_19877108646127
// MI455X (gfx1250) — matrix operand order repaired and checked by reading, not compiled, not run on hardware
//
#include <hip/hip_runtime.h>
#include <cstdint>
#include <cstddef>

typedef _Float16 f16;
typedef __attribute__((ext_vector_type(16))) _Float16 v16h;
typedef __attribute__((ext_vector_type(8)))  float    v8f;

// ---------------- model constants ----------------
#define DM    512      // d_model
#define DFF   2048     // d_ff
#define NBATCH 16
#define LSEQ  512
#define BLROWS (NBATCH*LSEQ)   // 8192
#define NH    8
#define DH    64
#define ZBH   (NBATCH*NH)      // 128
#define CIN   7
#define PREDL 256

// ---------------- convert f32 -> f16 (x4 vectorized) ----------------
__global__ void k_cvt16(const float4* __restrict__ s, f16* __restrict__ d, int n4) {
  int i = blockIdx.x * blockDim.x + threadIdx.x;
  if (i >= n4) return;
  float4 v = s[i];
  union { f16 h[4]; uint2 u; } t;
  t.h[0] = (f16)v.x; t.h[1] = (f16)v.y; t.h[2] = (f16)v.z; t.h[3] = (f16)v.w;
  *(uint2*)(d + (size_t)i * 4) = t.u;
}

// ---------------- pack weight: f32 [K][N] -> f16 [N][K] ----------------
__global__ void k_packT(const float* __restrict__ w, f16* __restrict__ wt, int K, int N) {
  int i = blockIdx.x * blockDim.x + threadIdx.x;
  if (i >= K * N) return;
  int n = i / K, k = i - n * K;
  wt[i] = (f16)w[(size_t)k * N + n];
}

// ---------------- embedding: circular conv3 + pos-enc + mark proj ----------------
__global__ void k_embed(const float* __restrict__ x, const float* __restrict__ xm,
                        const float* __restrict__ convw, const float* __restrict__ tw,
                        const float* __restrict__ tb, float* __restrict__ out, int L) {
  int row = blockIdx.x;              // b*L + l
  int b = row / L, l = row - b * L;
  int t = threadIdx.x;               // 256 threads
  __shared__ float sx[3 * CIN];
  __shared__ float sm[4];
  if (t < 3 * CIN) {
    int w = t / CIN, c = t - w * CIN;
    int ls = l - 1 + w; ls = (ls + L) % L;
    sx[t] = x[((size_t)b * L + ls) * CIN + c];
  }
  if (t < 4) sm[t] = xm[((size_t)b * L + l) * 4 + t];
  __syncthreads();
  for (int d = t; d < DM; d += 256) {
    float acc = tb[d];
#pragma unroll
    for (int w = 0; w < 3; w++)
#pragma unroll
      for (int c = 0; c < CIN; c++)
        acc += sx[w * CIN + c] * convw[(w * CIN + c) * DM + d];
#pragma unroll
    for (int m = 0; m < 4; m++) acc += sm[m] * tw[m * DM + d];
    int i2 = d >> 1;
    float div = __expf(-(float)(2 * i2) * (9.210340371976184f / (float)DM));
    float ang = (float)l * div;
    acc += (d & 1) ? __cosf(ang) : __sinf(ang);
    out[(size_t)row * DM + d] = acc;
  }
}

// ---------------- fused (a [+ b]) -> LayerNorm ----------------
__global__ void k_addln(const float* __restrict__ a, const float* __restrict__ bsrc,
                        const float* __restrict__ g, const float* __restrict__ beta,
                        float* __restrict__ out, int hasB) {
  int row = blockIdx.x;
  int t = threadIdx.x;               // 256, 2 elems each (DM=512)
  const float* pa = a + (size_t)row * DM;
  const float* pb = bsrc + (size_t)row * DM;
  float v0 = pa[t]       + (hasB ? pb[t]       : 0.f);
  float v1 = pa[t + 256] + (hasB ? pb[t + 256] : 0.f);
  __shared__ float s1[256], s2[256];
  s1[t] = v0 + v1; s2[t] = v0 * v0 + v1 * v1;
  __syncthreads();
  for (int o = 128; o > 0; o >>= 1) {
    if (t < o) { s1[t] += s1[t + o]; s2[t] += s2[t + o]; }
    __syncthreads();
  }
  float mean = s1[0] * (1.f / DM);
  float var  = s2[0] * (1.f / DM) - mean * mean;
  float inv  = rsqrtf(var + 1e-5f);
  float* po = out + (size_t)row * DM;
  po[t]       = (v0 - mean) * inv * g[t]       + beta[t];
  po[t + 256] = (v1 - mean) * inv * g[t + 256] + beta[t + 256];
}

// ---------------- fused scale + (causal) softmax, f32 -> f16 ----------------
__global__ void k_softmax(const float* __restrict__ sc, f16* __restrict__ pr,
                          float scale, int causal) {
  int l = blockIdx.x;                // query row 0..511
  int z = blockIdx.y;                // batch*head
  const float* row = sc + ((size_t)z * LSEQ + l) * LSEQ;
  f16* orow = pr + ((size_t)z * LSEQ + l) * LSEQ;
  int t = threadIdx.x;               // 256, 2 cols each
  float v0 = row[t] * scale, v1 = row[t + 256] * scale;
  if (causal) { if (t > l) v0 = -1e30f; if (t + 256 > l) v1 = -1e30f; }
  __shared__ float s1[256];
  s1[t] = fmaxf(v0, v1);
  __syncthreads();
  for (int o = 128; o > 0; o >>= 1) { if (t < o) s1[t] = fmaxf(s1[t], s1[t + o]); __syncthreads(); }
  float mx = s1[0];
  __syncthreads();
  float e0 = __expf(v0 - mx), e1 = __expf(v1 - mx);
  s1[t] = e0 + e1;
  __syncthreads();
  for (int o = 128; o > 0; o >>= 1) { if (t < o) s1[t] += s1[t + o]; __syncthreads(); }
  float inv = 1.f / s1[0];
  orow[t] = (f16)(e0 * inv);
  orow[t + 256] = (f16)(e1 * inv);
}

// ---------------- final projection (N=7) + slice last PRED_LEN ----------------
__global__ void k_proj(const float* __restrict__ x, const float* __restrict__ w,
                       const float* __restrict__ b, float* __restrict__ out) {
  int i = blockIdx.x * blockDim.x + threadIdx.x;
  if (i >= NBATCH * PREDL * CIN) return;
  int c = i % CIN; int t = i / CIN; int lp = t % PREDL; int bb = t / PREDL;
  const float* row = x + ((size_t)bb * LSEQ + (LSEQ - PREDL) + lp) * DM;
  float acc = b[c];
  for (int k = 0; k < DM; k++) acc += row[k] * w[k * CIN + c];
  out[i] = acc;
}

// ---------------- WMMA GEMM with async LDS double buffering ----------------
// C[M][N] = A[M][K] * B + bias, f16 in, f32 accumulate, v_wmma_f32_16x16x32_f16.
// BT=1: B stored [N][K] (pre-transposed weights / K-matrix for QK^T) -> async fill.
// BT=0: B stored [K][N] (V for P*V); transposed via registers while staging.
// A tile always async-filled (GLOBAL_LOAD_ASYNC_TO_LDS_B128, ASYNCcnt pipelined).
// Batched over blockIdx.z with two-level strides: off = (z>>3)*s1 + (z&7)*s2.
// act==1: exact GELU epilogue.
#define LDS_STRIDE 40   // halves; 80B row pitch (16B aligned)
#define TILE_HALVES (128 * LDS_STRIDE)
template <int BT>
__global__ __launch_bounds__(256) void k_gemm(
    const f16* __restrict__ A, int lda, long long sA1, long long sA2,
    const f16* __restrict__ Bm, int ldb, long long sB1, long long sB2,
    float* __restrict__ C, int ldc, long long sC1, long long sC2,
    const float* __restrict__ bias, int M, int N, int K, int act) {
  __shared__ __align__(16) f16 As[2][TILE_HALVES];
  __shared__ __align__(16) f16 Bs[2][TILE_HALVES];
  int z = blockIdx.z;
  A  += (long long)(z >> 3) * sA1 + (long long)(z & 7) * sA2;
  Bm += (long long)(z >> 3) * sB1 + (long long)(z & 7) * sB2;
  C  += (long long)(z >> 3) * sC1 + (long long)(z & 7) * sC2;
  int n0 = blockIdx.x * 128, m0 = blockIdx.y * 128;
  int tid = threadIdx.x;
  int lane = tid & 31, wave = tid >> 5;
  int wm = wave & 3, wn = wave >> 2;           // wave tile: 32 rows x 64 cols
  int lh = lane & 15, hsel = lane >> 4;
  bool fullN = (n0 + 128 <= N);                // true for every BT=1 call we make
  uint32_t asBase = (uint32_t)(uintptr_t)&As[0][0];  // LDS byte offset (addr[31:0])
  uint32_t bsBase = (uint32_t)(uintptr_t)&Bs[0][0];

  v8f acc[2][4];
#pragma unroll
  for (int i = 0; i < 2; i++)
#pragma unroll
    for (int j = 0; j < 4; j++)
#pragma unroll
      for (int e = 0; e < 8; e++) acc[i][j][e] = 0.f;

  // stage one 128x32 K-tile pair into buffer `buf`
  auto stage = [&](int buf, int k0) {
    uint32_t aB = asBase + (uint32_t)buf * (TILE_HALVES * 2);
    uint32_t bB = bsBase + (uint32_t)buf * (TILE_HALVES * 2);
    // ---- A tile: 512 async 16B chunks across 256 threads ----
#pragma unroll
    for (int it = 0; it < 2; it++) {
      int q = tid + it * 256;
      int r = q >> 2, qd = q & 3;
      unsigned long long g =
          (unsigned long long)(uintptr_t)(A + (long long)(m0 + r) * lda + k0 + qd * 8);
      uint32_t l = aB + (uint32_t)(r * LDS_STRIDE + qd * 8) * 2;
      asm volatile("global_load_async_to_lds_b128 %0, %1, off" :: "v"(l), "v"(g) : "memory");
    }
    if (BT && fullN) {
      // ---- B tile [N][K]: async, same pattern ----
#pragma unroll
      for (int it = 0; it < 2; it++) {
        int q = tid + it * 256;
        int r = q >> 2, qd = q & 3;
        unsigned long long g =
            (unsigned long long)(uintptr_t)(Bm + (long long)(n0 + r) * ldb + k0 + qd * 8);
        uint32_t l = bB + (uint32_t)(r * LDS_STRIDE + qd * 8) * 2;
        asm volatile("global_load_async_to_lds_b128 %0, %1, off" :: "v"(l), "v"(g) : "memory");
      }
    } else if (BT) {
      // guarded fallback (N not multiple of 128) via registers
#pragma unroll
      for (int it = 0; it < 2; it++) {
        int q = tid + it * 256;
        int r = q >> 2, qd = q & 3;
        uint4 val; val.x = val.y = val.z = val.w = 0u;
        if (n0 + r < N)
          val = *(const uint4*)(Bm + (long long)(n0 + r) * ldb + k0 + qd * 8);
        *(uint4*)(&Bs[buf][r * LDS_STRIDE + qd * 8]) = val;
      }
    } else {
      // ---- B tile [K][N]: transpose while staging (P*V path, N=64) ----
#pragma unroll
      for (int it = 0; it < 2; it++) {
        int q = tid + it * 256;
        int r = q >> 4, qd = q & 15;           // r: k-row 0..31, qd: n-octet
        int nn = n0 + qd * 8;
        union { uint4 u; f16 h[8]; } t;
        if (nn + 8 <= N)
          t.u = *(const uint4*)(Bm + (long long)(k0 + r) * ldb + nn);
        else {
#pragma unroll
          for (int jj = 0; jj < 8; jj++) t.h[jj] = (f16)0.f;
        }
#pragma unroll
        for (int jj = 0; jj < 8; jj++)
          Bs[buf][(qd * 8 + jj) * LDS_STRIDE + r] = t.h[jj];
      }
    }
  };

  const int TILES = K >> 5;
  stage(0, 0);                                  // prologue fill
  for (int t = 0; t < TILES; ++t) {
    int cur = t & 1;
    if (t + 1 < TILES) stage(1 - cur, (t + 1) << 5);   // prefetch next tile
    // wait for CURRENT tile's async fills (in-order completion; next tile's
    // APT ops may remain outstanding), then converge all waves.
    if (t + 1 < TILES) {
      if (BT && fullN) asm volatile("s_wait_asynccnt 0x4" ::: "memory");
      else             asm volatile("s_wait_asynccnt 0x2" ::: "memory");
    } else {
      asm volatile("s_wait_asynccnt 0x0" ::: "memory");
    }
    __syncthreads();

    // ---- fragments (ISA 7.12.2 layouts) ----
    union Frag { uint4 q[2]; v16h h; };
    Frag fa[2], fb[4];
#pragma unroll
    for (int i = 0; i < 2; i++) {
      const f16* base = &As[cur][(wm * 32 + i * 16 + lh) * LDS_STRIDE];
      // A 16x32: VGPR g<4 -> K=2g..2g+1 (+8 if lane>=16); g>=4 -> +16
      fa[i].q[0] = *(const uint4*)(base + hsel * 8);
      fa[i].q[1] = *(const uint4*)(base + 16 + hsel * 8);
    }
#pragma unroll
    for (int j = 0; j < 4; j++) {
      const f16* base = &Bs[cur][(wn * 64 + j * 16 + lh) * LDS_STRIDE];
      // B 32x16: same K order as A: VGPR g<4 -> K=2g..2g+1 (+8 if lane>=16); g>=4 -> +16
      fb[j].q[0] = *(const uint4*)(base + hsel * 8);
      fb[j].q[1] = *(const uint4*)(base + 16 + hsel * 8);
    }
#pragma unroll
    for (int i = 0; i < 2; i++)
#pragma unroll
      for (int j = 0; j < 4; j++)
        acc[i][j] = __builtin_amdgcn_wmma_f32_16x16x32_f16(
            false, fa[i].h, false, fb[j].h, (short)0, acc[i][j], false, false);
    __syncthreads();   // reads done before buffer `cur` is refilled next iter
  }

  // ---- epilogue: bias (+GELU), store f32 ----
#pragma unroll
  for (int j = 0; j < 4; j++) {
    int col = n0 + wn * 64 + j * 16 + lh;
    if (col >= N) continue;
    float bv = bias ? bias[col] : 0.f;
#pragma unroll
    for (int i = 0; i < 2; i++) {
      int rbase = m0 + wm * 32 + i * 16 + hsel * 8;
#pragma unroll
      for (int r = 0; r < 8; r++) {
        float v = acc[i][j][r] + bv;
        if (act == 1) v = 0.5f * v * (1.f + erff(v * 0.70710678118f));
        C[(long long)(rbase + r) * ldc + col] = v;
      }
    }
  }
}

// =======================================================================
extern "C" void kernel_launch(void* const* d_in, const int* in_sizes, int n_in,
                              void* d_out, int out_size, void* d_ws, size_t ws_size,
                              hipStream_t stream) {
  (void)in_sizes; (void)n_in; (void)out_size; (void)ws_size;
  // ---- input mapping (setup_inputs dict insertion order, recursively) ----
  const float* x_enc      = (const float*)d_in[0];
  const float* x_mark_enc = (const float*)d_in[1];
  const float* x_dec      = (const float*)d_in[2];
  const float* x_mark_dec = (const float*)d_in[3];
  auto P = [&](int i) -> const float* { return (const float*)d_in[i]; };
  // enc_emb: 4 conv_w, 5 temp_w, 6 temp_b ; dec_emb: 7,8,9
  // enc layer i base 10+16i: w1,b1,w2,b2,ln1g,ln1b,ln2g,ln2b,wq,bq,wk,bk,wv,bv,wo,bo
  // 58,59 enc_norm g,b
  // dec layer j base 60+26j: w1,b1,w2,b2,ln1g..ln3b(4..9),
  //   self wq..bo (10..17), cross wq..bo (18..25)
  // 112,113 dec_norm g,b ; 114 proj_w ; 115 proj_b

  // ---- workspace bump allocator ----
  char* ws = (char*)d_ws;
  size_t off = 0;
  auto alloc = [&](size_t elems, size_t esz) -> void* {
    off = (off + 255) & ~(size_t)255;
    void* p = ws + off;
    off += elems * esz;
    return p;
  };
  const size_t BLD = (size_t)BLROWS * DM;       // 4.19M
  const size_t SCN = (size_t)ZBH * LSEQ * LSEQ; // 33.55M

  float* enc  = (float*)alloc(BLD, 4);
  float* decb = (float*)alloc(BLD, 4);
  float* xb   = (float*)alloc(BLD, 4);
  float* tmp  = (float*)alloc(BLD, 4);
  float* ctx  = (float*)alloc(BLD, 4);
  float* qf   = (float*)alloc(BLD, 4);
  float* kf   = (float*)alloc(BLD, 4);
  float* vf   = (float*)alloc(BLD, 4);
  f16* x16    = (f16*)alloc(BLD, 2);
  f16* q16    = (f16*)alloc(BLD, 2);
  f16* k16    = (f16*)alloc(BLD, 2);
  f16* v16    = (f16*)alloc(BLD, 2);
  f16* enc16  = (f16*)alloc(BLD, 2);
  float* scores = (float*)alloc(SCN, 4);  // reused as FFN hidden f32
  float* hid    = scores;
  f16* probs    = (f16*)alloc(SCN, 2);    // reused as FFN hidden f16
  f16* h16      = probs;

  // ---- pack weights -> transposed f16 ----
  auto pack = [&](const float* w, f16* wt, int K, int N) {
    int n = K * N;
    k_packT<<<(n + 255) / 256, 256, 0, stream>>>(w, wt, K, N);
  };
  f16 *e_wq[3], *e_wk[3], *e_wv[3], *e_wo[3], *e_w1[3], *e_w2[3];
  for (int i = 0; i < 3; i++) {
    int e = 10 + 16 * i;
    e_wq[i] = (f16*)alloc((size_t)DM * DM, 2);  pack(P(e + 8),  e_wq[i], DM, DM);
    e_wk[i] = (f16*)alloc((size_t)DM * DM, 2);  pack(P(e + 10), e_wk[i], DM, DM);
    e_wv[i] = (f16*)alloc((size_t)DM * DM, 2);  pack(P(e + 12), e_wv[i], DM, DM);
    e_wo[i] = (f16*)alloc((size_t)DM * DM, 2);  pack(P(e + 14), e_wo[i], DM, DM);
    e_w1[i] = (f16*)alloc((size_t)DM * DFF, 2); pack(P(e + 0),  e_w1[i], DM, DFF);
    e_w2[i] = (f16*)alloc((size_t)DM * DFF, 2); pack(P(e + 2),  e_w2[i], DFF, DM);
  }
  f16 *d_sq[2], *d_sk[2], *d_sv[2], *d_so[2], *d_cq[2], *d_ck[2], *d_cv[2], *d_co[2],
      *d_w1[2], *d_w2[2];
  for (int j = 0; j < 2; j++) {
    int d0 = 60 + 26 * j;
    d_sq[j] = (f16*)alloc((size_t)DM * DM, 2);  pack(P(d0 + 10), d_sq[j], DM, DM);
    d_sk[j] = (f16*)alloc((size_t)DM * DM, 2);  pack(P(d0 + 12), d_sk[j], DM, DM);
    d_sv[j] = (f16*)alloc((size_t)DM * DM, 2);  pack(P(d0 + 14), d_sv[j], DM, DM);
    d_so[j] = (f16*)alloc((size_t)DM * DM, 2);  pack(P(d0 + 16), d_so[j], DM, DM);
    d_cq[j] = (f16*)alloc((size_t)DM * DM, 2);  pack(P(d0 + 18), d_cq[j], DM, DM);
    d_ck[j] = (f16*)alloc((size_t)DM * DM, 2);  pack(P(d0 + 20), d_ck[j], DM, DM);
    d_cv[j] = (f16*)alloc((size_t)DM * DM, 2);  pack(P(d0 + 22), d_cv[j], DM, DM);
    d_co[j] = (f16*)alloc((size_t)DM * DM, 2);  pack(P(d0 + 24), d_co[j], DM, DM);
    d_w1[j] = (f16*)alloc((size_t)DM * DFF, 2); pack(P(d0 + 0), d_w1[j], DM, DFF);
    d_w2[j] = (f16*)alloc((size_t)DM * DFF, 2); pack(P(d0 + 2), d_w2[j], DFF, DM);
  }

  // ---- helpers ----
  auto cvt = [&](const float* s, f16* d, size_t n) {
    int n4 = (int)(n / 4);
    k_cvt16<<<(n4 + 255) / 256, 256, 0, stream>>>((const float4*)s, d, n4);
  };
  auto gemm = [&](int BT, const f16* Ap, int lda, long long sA1, long long sA2,
                  const f16* Bp, int ldb, long long sB1, long long sB2,
                  float* Cp, int ldc, long long sC1, long long sC2,
                  const float* bias, int M, int N, int K, int act, int Z) {
    dim3 g((N + 127) / 128, (M + 127) / 128, Z), blk(256);
    if (BT)
      hipLaunchKernelGGL((k_gemm<1>), g, blk, 0, stream, Ap, lda, sA1, sA2,
                         Bp, ldb, sB1, sB2, Cp, ldc, sC1, sC2, bias, M, N, K, act);
    else
      hipLaunchKernelGGL((k_gemm<0>), g, blk, 0, stream, Ap, lda, sA1, sA2,
                         Bp, ldb, sB1, sB2, Cp, ldc, sC1, sC2, bias, M, N, K, act);
  };
  auto addln = [&](const float* a, const float* b, const float* g, const float* be,
                   float* o, int hasB) {
    k_addln<<<BLROWS, 256, 0, stream>>>(a, b ? b : a, g, be, o, hasB);
  };
  const long long SBH = (long long)LSEQ * DM;     // per-batch row-block stride
  const long long SSC = (long long)LSEQ * LSEQ;   // per-head score stride
  auto attention = [&](float* ctxp, int causal) {
    // scores[z,l,s] = Q[z,l,:] . K[z,s,:]   (BT=1, K stored [s][d])
    gemm(1, q16, DM, SBH, DH, k16, DM, SBH, DH,
         scores, LSEQ, (long long)NH * SSC, SSC,
         nullptr, LSEQ, LSEQ, DH, 0, ZBH);
    k_softmax<<<dim3(LSEQ, ZBH), 256, 0, stream>>>(scores, probs, 0.125f, causal);
    // ctx[b,l,h*64+d] = P[z,l,:] . V[z,:,d]  (BT=0, V stored [s][d])
    gemm(0, probs, LSEQ, (long long)NH * SSC, SSC, v16, DM, SBH, DH,
         ctxp, DM, SBH, DH, nullptr, LSEQ, DH, LSEQ, 0, ZBH);
  };

  // ================= encoder =================
  k_embed<<<BLROWS, 256, 0, stream>>>(x_enc, x_mark_enc, P(4), P(5), P(6), enc, LSEQ);
  for (int i = 0; i < 3; i++) {
    int e = 10 + 16 * i;
    cvt(enc, x16, BLD);
    gemm(1, x16, DM, 0, 0, e_wq[i], DM, 0, 0, qf, DM, 0, 0, P(e + 9),  BLROWS, DM, DM, 0, 1);
    gemm(1, x16, DM, 0, 0, e_wk[i], DM, 0, 0, kf, DM, 0, 0, P(e + 11), BLROWS, DM, DM, 0, 1);
    gemm(1, x16, DM, 0, 0, e_wv[i], DM, 0, 0, vf, DM, 0, 0, P(e + 13), BLROWS, DM, DM, 0, 1);
    cvt(qf, q16, BLD); cvt(kf, k16, BLD); cvt(vf, v16, BLD);
    attention(ctx, 0);
    cvt(ctx, x16, BLD);
    gemm(1, x16, DM, 0, 0, e_wo[i], DM, 0, 0, tmp, DM, 0, 0, P(e + 15), BLROWS, DM, DM, 0, 1);
    addln(enc, tmp, P(e + 4), P(e + 5), xb, 1);
    cvt(xb, x16, BLD);
    gemm(1, x16, DM, 0, 0, e_w1[i], DM, 0, 0, hid, DFF, 0, 0, P(e + 1), BLROWS, DFF, DM, 1, 1);
    cvt(hid, h16, (size_t)BLROWS * DFF);
    gemm(1, h16, DFF, 0, 0, e_w2[i], DFF, 0, 0, tmp, DM, 0, 0, P(e + 3), BLROWS, DM, DFF, 0, 1);
    addln(xb, tmp, P(e + 6), P(e + 7), enc, 1);
  }
  addln(enc, nullptr, P(58), P(59), enc, 0);
  cvt(enc, enc16, BLD);

  // ================= decoder =================
  k_embed<<<BLROWS, 256, 0, stream>>>(x_dec, x_mark_dec, P(7), P(8), P(9), decb, LSEQ);
  for (int j = 0; j < 2; j++) {
    int d0 = 60 + 26 * j;
    // self attention (causal)
    cvt(decb, x16, BLD);
    gemm(1, x16, DM, 0, 0, d_sq[j], DM, 0, 0, qf, DM, 0, 0, P(d0 + 11), BLROWS, DM, DM, 0, 1);
    gemm(1, x16, DM, 0, 0, d_sk[j], DM, 0, 0, kf, DM, 0, 0, P(d0 + 13), BLROWS, DM, DM, 0, 1);
    gemm(1, x16, DM, 0, 0, d_sv[j], DM, 0, 0, vf, DM, 0, 0, P(d0 + 15), BLROWS, DM, DM, 0, 1);
    cvt(qf, q16, BLD); cvt(kf, k16, BLD); cvt(vf, v16, BLD);
    attention(ctx, 1);
    cvt(ctx, x16, BLD);
    gemm(1, x16, DM, 0, 0, d_so[j], DM, 0, 0, tmp, DM, 0, 0, P(d0 + 17), BLROWS, DM, DM, 0, 1);
    addln(decb, tmp, P(d0 + 4), P(d0 + 5), xb, 1);
    // cross attention
    cvt(xb, x16, BLD);
    gemm(1, x16,   DM, 0, 0, d_cq[j], DM, 0, 0, qf, DM, 0, 0, P(d0 + 19), BLROWS, DM, DM, 0, 1);
    gemm(1, enc16, DM, 0, 0, d_ck[j], DM, 0, 0, kf, DM, 0, 0, P(d0 + 21), BLROWS, DM, DM, 0, 1);
    gemm(1, enc16, DM, 0, 0, d_cv[j], DM, 0, 0, vf, DM, 0, 0, P(d0 + 23), BLROWS, DM, DM, 0, 1);
    cvt(qf, q16, BLD); cvt(kf, k16, BLD); cvt(vf, v16, BLD);
    attention(ctx, 0);
    cvt(ctx, x16, BLD);
    gemm(1, x16, DM, 0, 0, d_co[j], DM, 0, 0, tmp, DM, 0, 0, P(d0 + 25), BLROWS, DM, DM, 0, 1);
    addln(xb, tmp, P(d0 + 6), P(d0 + 7), ctx, 1);   // x2 lives in ctx
    // FFN
    cvt(ctx, x16, BLD);
    gemm(1, x16, DM, 0, 0, d_w1[j], DM, 0, 0, hid, DFF, 0, 0, P(d0 + 1), BLROWS, DFF, DM, 1, 1);
    cvt(hid, h16, (size_t)BLROWS * DFF);
    gemm(1, h16, DFF, 0, 0, d_w2[j], DFF, 0, 0, tmp, DM, 0, 0, P(d0 + 3), BLROWS, DM, DFF, 0, 1);
    addln(ctx, tmp, P(d0 + 8), P(d0 + 9), decb, 1);
  }
  addln(decb, nullptr, P(112), P(113), decb, 0);

  // ================= output projection + slice =================
  k_proj<<<(NBATCH * PREDL * CIN + 255) / 256, 256, 0, stream>>>(
      decb, P(114), P(115), (float*)d_out);
}
